// Model_14027363188804
// MI455X (gfx1250) — matrix operand order repaired and checked by reading, not compiled, not run on hardware
//
#include <hip/hip_runtime.h>
#include <cstdint>
#include <cstddef>

typedef __attribute__((ext_vector_type(16))) _Float16 v16h;
typedef __attribute__((ext_vector_type(8)))  _Float16 v8h;
typedef __attribute__((ext_vector_type(8)))  float    v8f;

static const int PLANES[7] = {32, 64, 96, 128, 160, 192, 224};

#define MAX_KT 14                    // Cin up to 448
#define RS_PAD 8                     // +16B per LDS row to destagger banks

// ---------------------------------------------------------------------------
// Kernels
// ---------------------------------------------------------------------------

// Repack f32 weights [K][Cin][Cout] into WMMA B-fragment order (f16):
// P[(((k*KT+kt)*NT+nt)*32 + lane)*16 + j] = W[k][kt*32 + (lane>>4)*16 + j][nt*16 + (lane&15)]
__global__ void k_pack_w(const float* __restrict__ W, _Float16* __restrict__ P,
                         int K, int Cin, int Cout, int KT, int NT) {
  long long total = (long long)K * KT * NT * 512;
  long long i = (long long)blockIdx.x * blockDim.x + threadIdx.x;
  if (i >= total) return;
  int j    = (int)(i & 15);
  int lane = (int)((i >> 4) & 31);
  long long t = i >> 9;                 // k*KT*NT + kt*NT + nt
  int nt = (int)(t % NT); t /= NT;
  int kt = (int)(t % KT);
  int k  = (int)(t / KT);
  int cin  = kt * 32 + (lane >> 4) * 16 + j;
  int cout = nt * 16 + (lane & 15);
  float v = 0.0f;
  if (cin < Cin && cout < Cout) v = W[((size_t)k * Cin + cin) * Cout + cout];
  P[i] = (_Float16)v;
}

// f32 [N][C] -> f16 [N][Cpad] with zero column padding.
__global__ void k_cvt16(const float* __restrict__ X, int N, int C, int Cpad,
                        _Float16* __restrict__ Y) {
  long long i = (long long)blockIdx.x * blockDim.x + threadIdx.x;
  long long total = (long long)N * Cpad;
  if (i >= total) return;
  int c = (int)(i % Cpad);
  long long n = i / Cpad;
  float v = (c < C) ? X[n * C + c] : 0.0f;
  Y[i] = (_Float16)v;
}

__global__ void k_zero_f32(float* __restrict__ p, long long n) {
  long long i = (long long)blockIdx.x * blockDim.x + threadIdx.x;
  if (i < n) p[i] = 0.0f;
}

// Per-channel mean + rsqrt(var+eps). One block of 256 threads per channel.
__global__ void k_bn_stats(const float* __restrict__ X, int N, int C,
                           float* __restrict__ mu, float* __restrict__ rs) {
  int c = blockIdx.x;
  float s = 0.0f, s2 = 0.0f;
  for (int n = threadIdx.x; n < N; n += 256) {
    float v = X[(size_t)n * C + c];
    s += v; s2 += v * v;
  }
  __shared__ float sh0[256];
  __shared__ float sh1[256];
  sh0[threadIdx.x] = s; sh1[threadIdx.x] = s2;
  __syncthreads();
  for (int o = 128; o > 0; o >>= 1) {
    if ((int)threadIdx.x < o) {
      sh0[threadIdx.x] += sh0[threadIdx.x + o];
      sh1[threadIdx.x] += sh1[threadIdx.x + o];
    }
    __syncthreads();
  }
  if (threadIdx.x == 0) {
    float m   = sh0[0] / (float)N;
    float var = sh1[0] / (float)N - m * m;
    mu[c] = m;
    rs[c] = rsqrtf(var + 1e-4f);
  }
}

// y = relu((x-mu)*rs*g + b); optional f16 (padded) and/or f32 outputs.
__global__ void k_bn_apply(const float* __restrict__ X, int N, int C, int Cpad,
                           const float* __restrict__ mu, const float* __restrict__ rs,
                           const float* __restrict__ g,  const float* __restrict__ b,
                           _Float16* __restrict__ Y16, float* __restrict__ Yf) {
  long long i = (long long)blockIdx.x * blockDim.x + threadIdx.x;
  long long total = (long long)N * Cpad;
  if (i >= total) return;
  int c = (int)(i % Cpad);
  long long n = i / Cpad;
  float v = 0.0f;
  if (c < C) {
    float x = X[n * C + c];
    v = (x - mu[c]) * rs[c] * g[c] + b[c];
    v = v > 0.0f ? v : 0.0f;
  }
  if (Y16) Y16[n * Cpad + c] = (_Float16)v;
  if (Yf && c < C) Yf[n * C + c] = v;
}

// Submanifold conv / GEMM via WMMA, LDS-staged A-tile, 4 waves per workgroup.
// One workgroup per 16-site row tile. Per offset k: the gathered 16 x Cin A-tile
// is staged into LDS once (async global->LDS DMA; zero rows for missing nbrs),
// then each wave computes up to 4 column tiles (nt = wave, wave+4, ...),
// amortizing A-fragment LDS reads across its slots.
__global__ void __launch_bounds__(128)
k_smconv_wmma(const _Float16* __restrict__ Xh, int ldx,
              const int* __restrict__ nbr, int K,
              const _Float16* __restrict__ Wp, int KT, int NT,
              int N, int Cout,
              float* __restrict__ out, int ldo, int ocol,
              const float* __restrict__ add, int lda, int acol,
              const float* __restrict__ bias) {
  const int mbase = blockIdx.x * 16;
  const int tid   = threadIdx.x;
  const int wave  = tid >> 5;
  const int lane  = tid & 31;
  const int half  = lane >> 4;
  const int l16   = lane & 15;
  const int RS    = KT * 32 + RS_PAD;      // LDS row stride in halves

  __shared__ _Float16 sA[16 * (MAX_KT * 32 + RS_PAD)];
  __shared__ int sidx[16];
  __shared__ int sflag;

  v8f acc[4];
#pragma unroll
  for (int s = 0; s < 4; ++s)
#pragma unroll
    for (int r = 0; r < 8; ++r) acc[s][r] = 0.0f;

  union AU { v16h v; v8h h[2]; };
  const int chunks = 16 * KT * 2;          // 16B chunks to stage per offset

  for (int k = 0; k < K; ++k) {
    // wave 0 gathers the 16 neighbor indices + workgroup-uniform skip flag
    if (tid < 32) {
      int row  = tid & 15;
      int site = mbase + row;
      int idx;
      if (nbr) idx = (site < N) ? nbr[(size_t)site * K + k] : -1;
      else     idx = (site < N) ? site : -1;
      sidx[row] = idx;
      unsigned long long bal = __ballot(idx >= 0);
      if (tid == 0) sflag = (bal != 0ull) ? 1 : 0;
    }
    __syncthreads();                       // publish sidx/sflag; fence prev consume
    const bool go = (sflag != 0);
    if (go) {
      // stage gathered A-tile into LDS (async DMA; ds_store zeros for holes)
      for (int ci = tid; ci < chunks; ci += 128) {
        int row  = ci / (KT * 2);
        int part = ci - row * (KT * 2);
        int idx  = sidx[row];
        _Float16* dst = &sA[(size_t)row * RS + part * 8];
        if (idx >= 0) {
          unsigned lds = (unsigned)(uintptr_t)dst;   // low 32b of generic = LDS offset
          unsigned long long ga =
              (unsigned long long)(uintptr_t)(Xh + (size_t)idx * (size_t)ldx + part * 8);
          asm volatile("global_load_async_to_lds_b128 %0, %1, off"
                       :: "v"(lds), "v"(ga) : "memory");
        } else {
          v8h z;
#pragma unroll
          for (int j = 0; j < 8; ++j) z[j] = (_Float16)0.0f;
          *reinterpret_cast<v8h*>(dst) = z;
        }
      }
      asm volatile("s_wait_asynccnt 0x0" ::: "memory");
    }
    __syncthreads();                       // A-tile visible to all waves
    if (go) {
      const _Float16* wk = Wp + (size_t)k * KT * NT * 512;
      if (k + 1 < K)                       // global_prefetch_b8 on next offset's B
        __builtin_prefetch(Wp + (size_t)(k + 1) * KT * NT * 512 + lane * 16, 0, 1);
      for (int kt = 0; kt < KT; ++kt) {
        AU a;                              // A fragment shared by all column slots
        const _Float16* ap = &sA[(size_t)l16 * RS + kt * 32 + half * 16];
        a.h[0] = *reinterpret_cast<const v8h*>(ap);
        a.h[1] = *reinterpret_cast<const v8h*>(ap + 8);
#pragma unroll
        for (int s = 0; s < 4; ++s) {
          int nt = wave + 4 * s;
          if (nt < NT) {
            v16h bfr = *reinterpret_cast<const v16h*>(
                wk + ((size_t)kt * NT + nt) * 512 + lane * 16);
            acc[s] = __builtin_amdgcn_wmma_f32_16x16x32_f16(false, a.v, false, bfr,
                                                            (short)0, acc[s], false, false);
          }
        }
      }
    }
  }

  // C layout: VGPR r -> M = half*8 + r, column = lane&15.
  const int rbase = mbase + half * 8;
#pragma unroll
  for (int s = 0; s < 4; ++s) {
    int nt = wave + 4 * s;
    if (nt >= NT) continue;
    int ncol = nt * 16 + l16;
    if (ncol >= Cout) continue;
#pragma unroll
    for (int r = 0; r < 8; ++r) {
      int row = rbase + r;
      if (row < N) {
        float v = acc[s][r];
        if (add)  v += add[(size_t)row * lda + acol + ncol];
        if (bias) v += bias[ncol];
        out[(size_t)row * ldo + ocol + ncol] = v;
      }
    }
  }
}

// Strided conv down: Y[inv[n]] += Xf[n] @ W[off[n]]   (atomic scatter-add)
__global__ void k_down_scatter(const float* __restrict__ Xf,
                               const int* __restrict__ inv, const int* __restrict__ koff,
                               const float* __restrict__ W,
                               int N, int Cin, int Cout, float* __restrict__ Y) {
  int n = blockIdx.x;
  if (n >= N) return;
  int k = koff[n], seg = inv[n];
  __shared__ float sx[256];
  for (int c = threadIdx.x; c < Cin; c += blockDim.x) sx[c] = Xf[(size_t)n * Cin + c];
  __syncthreads();
  const float* Wk = W + (size_t)k * Cin * Cout;
  for (int d = threadIdx.x; d < Cout; d += blockDim.x) {
    float s = 0.0f;
    for (int c = 0; c < Cin; ++c) s += sx[c] * Wk[(size_t)c * Cout + d];
    atomicAdd(&Y[(size_t)seg * Cout + d], s);
  }
}

// Deconv up: out[n, ocol + d] = Hc[inv[n]] @ W[off[n]]
__global__ void k_up_gather(const float* __restrict__ Hc,
                            const int* __restrict__ inv, const int* __restrict__ koff,
                            const float* __restrict__ W,
                            int N, int Cin, int Cout,
                            float* __restrict__ out, int ldo, int ocol) {
  int n = blockIdx.x;
  if (n >= N) return;
  int k = koff[n], src = inv[n];
  __shared__ float sx[256];
  for (int c = threadIdx.x; c < Cin; c += blockDim.x) sx[c] = Hc[(size_t)src * Cin + c];
  __syncthreads();
  const float* Wk = W + (size_t)k * Cin * Cout;
  for (int d = threadIdx.x; d < Cout; d += blockDim.x) {
    float s = 0.0f;
    for (int c = 0; c < Cin; ++c) s += sx[c] * Wk[(size_t)c * Cout + d];
    out[(size_t)n * ldo + ocol + d] = s;
  }
}

__global__ void k_copy_cols(const float* __restrict__ in, int N, int C,
                            float* __restrict__ out, int ldo, int ocol) {
  long long i = (long long)blockIdx.x * blockDim.x + threadIdx.x;
  long long total = (long long)N * C;
  if (i >= total) return;
  int c = (int)(i % C);
  long long n = i / C;
  out[n * ldo + ocol + c] = in[i];
}

// ---------------------------------------------------------------------------
// Host orchestration
// ---------------------------------------------------------------------------

// Input flattening order assumption: jax pytree flatten (dict keys sorted, lists
// in order). Top: feats, params..., rb...
static inline int ix_bd(int l)   { return 1 + l; }
static inline int ix_bf()        { return 7; }
static inline int ix_bu(int l)   { return 8 + l; }
static inline int ix_conv0()     { return 14; }
// down block fields (sorted): 0=b1 1=b2 2=g1 3=g2 4=w1 5=w2
static inline int ix_down(int l, int r, int f) { return 15 + (l * 2 + r) * 6 + f; }
static inline int ix_gd(int l)   { return 99 + l; }
static inline int ix_gf()        { return 105; }
static inline int ix_gu(int l)   { return 106 + l; }
static inline int ix_lb()        { return 112; }
static inline int ix_lw()        { return 113; }
// up r0 fields (sorted): 0=b1 1=b2 2=g1 3=g2 4=nin 5=w1 6=w2 ; r1: 0..5 = b1,b2,g1,g2,w1,w2
static inline int ix_up(int l, int r, int f) { return 114 + l * 13 + (r == 0 ? f : 7 + f); }
static inline int ix_wd(int l)   { return 192 + l; }
static inline int ix_wu(int l)   { return 198 + l; }
static inline int ix_inv(int l)  { return 204 + l; }
static inline int ix_nbr(int l)  { return 210 + l; }
static inline int ix_off(int l)  { return 217 + l; }

static inline int pad32(int c) { return (c + 31) & ~31; }

struct Bump {
  char*  base;
  size_t cur;
  void* take(size_t bytes) {
    void* p = base + cur;
    cur = (cur + bytes + 255) & ~(size_t)255;
    return p;
  }
};

struct Pack { const _Float16* p; int KT, NT; };

struct Ctx {
  hipStream_t stream;
  void* const* din;
  const int* nbr[7];
  const int* inv[6];
  const int* off[6];
  int Nl[7];
  float* X[7];
  float* XC;          // concat buffer (N x 2C, compact per level)
  float* H;           // f32 scratch (N0 x 448)
  float* SC;          // NiN shortcut (N0 x 224)
  _Float16* T16;      // bn-relu f16 (N0 x 448 padded)
  _Float16* T16b;     // raw-cvt f16 (N0 x 448 padded)
  float* mu;
  float* rs;
  Pack conv0, lw;
  Pack dw[7][2][2];
  Pack unin[6];
  Pack uw[6][2][2];
};

static inline const float* fP(const Ctx& c, int i) { return (const float*)c.din[i]; }
static inline const int*   fI(const Ctx& c, int i) { return (const int*)c.din[i]; }

static Pack pack_w(Ctx& c, Bump& ws, const float* W, int K, int Cin, int Cout) {
  Pack p;
  p.KT = (Cin + 31) / 32;
  p.NT = (Cout + 15) / 16;
  long long elems = (long long)K * p.KT * p.NT * 512;
  _Float16* dst = (_Float16*)ws.take((size_t)elems * sizeof(_Float16));
  k_pack_w<<<(unsigned)((elems + 255) / 256), 256, 0, c.stream>>>(W, dst, K, Cin, Cout, p.KT, p.NT);
  p.p = dst;
  return p;
}

static void smconv(Ctx& c, const _Float16* Xh, int ldx, const int* nbr, int K,
                   const Pack& w, int N, int Cout,
                   float* out, int ldo, int ocol,
                   const float* add, int lda, int acol, const float* bias) {
  dim3 grid((N + 15) / 16);
  k_smconv_wmma<<<grid, 128, 0, c.stream>>>(Xh, ldx, nbr, K, w.p, w.KT, w.NT,
                                            N, Cout, out, ldo, ocol, add, lda, acol, bias);
}

static void bn(Ctx& c, const float* X, int N, int C,
               const float* g, const float* b,
               _Float16* y16, int Cpad, float* yf) {
  k_bn_stats<<<C, 256, 0, c.stream>>>(X, N, C, c.mu, c.rs);
  long long total = (long long)N * Cpad;
  k_bn_apply<<<(unsigned)((total + 255) / 256), 256, 0, c.stream>>>(
      X, N, C, Cpad, c.mu, c.rs, g, b, y16, yf);
}

// Pre-activation residual block: out = sc + smconv(bnrelu(smconv(bnrelu(x))))
static void run_block(Ctx& c, const float* xin, int a, float* xout, int b_,
                      const float* g1, const float* b1, const Pack& w1,
                      const float* g2, const float* b2, const Pack& w2,
                      const Pack* nin, const int* nbr, int N) {
  const int apad = pad32(a), bpad = pad32(b_);
  bn(c, xin, N, a, g1, b1, c.T16, apad, nullptr);
  const float* add; int lda;
  if (nin) {
    long long total = (long long)N * apad;
    k_cvt16<<<(unsigned)((total + 255) / 256), 256, 0, c.stream>>>(xin, N, a, apad, c.T16b);
    smconv(c, c.T16b, apad, nullptr, 1, *nin, N, b_, c.SC, b_, 0, nullptr, 0, 0, nullptr);
    add = c.SC; lda = b_;
  } else {
    add = xin; lda = a;    // a == b_, identity shortcut
  }
  smconv(c, c.T16, apad, nbr, 27, w1, N, b_, c.H, b_, 0, nullptr, 0, 0, nullptr);
  bn(c, c.H, N, b_, g2, b2, c.T16, bpad, nullptr);
  smconv(c, c.T16, bpad, nbr, 27, w2, N, b_, xout, b_, 0, add, lda, 0, nullptr);
}

static void unet(Ctx& c, int l) {
  const int N = c.Nl[l], C = PLANES[l];
  const int* nbr = c.nbr[l];
  for (int r = 0; r < 2; ++r) {
    run_block(c, c.X[l], C, c.X[l], C,
              fP(c, ix_down(l, r, 2)), fP(c, ix_down(l, r, 0)), c.dw[l][r][0],
              fP(c, ix_down(l, r, 3)), fP(c, ix_down(l, r, 1)), c.dw[l][r][1],
              nullptr, nbr, N);
  }
  if (l == 6) return;
  const int Nc = c.Nl[l + 1], Cn = PLANES[l + 1];

  // --- down transition: bnrelu -> per-corner GEMV -> scatter-add ---
  bn(c, c.X[l], N, C, fP(c, ix_gd(l)), fP(c, ix_bd(l)), nullptr, C, c.H);
  long long zn = (long long)Nc * Cn;
  k_zero_f32<<<(unsigned)((zn + 255) / 256), 256, 0, c.stream>>>(c.X[l + 1], zn);
  k_down_scatter<<<N, 128, 0, c.stream>>>(c.H, c.inv[l], c.off[l],
                                          fP(c, ix_wd(l)), N, C, Cn, c.X[l + 1]);
  unet(c, l + 1);

  // --- up transition: bnrelu -> gather + per-corner GEMV into concat cols ---
  bn(c, c.X[l + 1], Nc, Cn, fP(c, ix_gu(l)), fP(c, ix_bu(l)), nullptr, Cn, c.H);
  k_up_gather<<<N, 128, 0, c.stream>>>(c.H, c.inv[l], c.off[l],
                                       fP(c, ix_wu(l)), N, Cn, C, c.XC, 2 * C, C);
  long long cn = (long long)N * C;
  k_copy_cols<<<(unsigned)((cn + 255) / 256), 256, 0, c.stream>>>(c.X[l], N, C, c.XC, 2 * C, 0);

  // --- up blocks: r0 has NiN shortcut (2C -> C), r1 identity (C -> C) ---
  run_block(c, c.XC, 2 * C, c.X[l], C,
            fP(c, ix_up(l, 0, 2)), fP(c, ix_up(l, 0, 0)), c.uw[l][0][0],
            fP(c, ix_up(l, 0, 3)), fP(c, ix_up(l, 0, 1)), c.uw[l][0][1],
            &c.unin[l], nbr, N);
  run_block(c, c.X[l], C, c.X[l], C,
            fP(c, ix_up(l, 1, 2)), fP(c, ix_up(l, 1, 0)), c.uw[l][1][0],
            fP(c, ix_up(l, 1, 3)), fP(c, ix_up(l, 1, 1)), c.uw[l][1][1],
            nullptr, nbr, N);
}

extern "C" void kernel_launch(void* const* d_in, const int* in_sizes, int n_in,
                              void* d_out, int out_size, void* d_ws, size_t ws_size,
                              hipStream_t stream) {
  (void)n_in; (void)out_size; (void)ws_size;
  Ctx c;
  c.stream = stream;
  c.din = d_in;
  for (int l = 0; l < 7; ++l) {
    c.nbr[l] = fI(c, ix_nbr(l));
    c.Nl[l]  = in_sizes[ix_nbr(l)] / 27;
  }
  for (int l = 0; l < 6; ++l) {
    c.inv[l] = fI(c, ix_inv(l));
    c.off[l] = fI(c, ix_off(l));
  }
  const int N0 = c.Nl[0];

  Bump ws{(char*)d_ws, 0};
  for (int l = 0; l < 7; ++l)
    c.X[l] = (float*)ws.take((size_t)c.Nl[l] * PLANES[l] * sizeof(float));
  c.XC   = (float*)ws.take((size_t)N0 * 448 * sizeof(float));
  c.H    = (float*)ws.take((size_t)N0 * 448 * sizeof(float));
  c.SC   = (float*)ws.take((size_t)N0 * 224 * sizeof(float));
  c.T16  = (_Float16*)ws.take((size_t)N0 * 448 * sizeof(_Float16));
  c.T16b = (_Float16*)ws.take((size_t)N0 * 448 * sizeof(_Float16));
  c.mu   = (float*)ws.take(512 * sizeof(float));
  c.rs   = (float*)ws.take(512 * sizeof(float));

  // ---- pack all WMMA weights (f32 -> f16 B-fragment layout) ----
  c.conv0 = pack_w(c, ws, fP(c, ix_conv0()), 27, 3, PLANES[0]);
  for (int l = 0; l < 7; ++l) {
    int C = PLANES[l];
    for (int r = 0; r < 2; ++r) {
      c.dw[l][r][0] = pack_w(c, ws, fP(c, ix_down(l, r, 4)), 27, C, C);
      c.dw[l][r][1] = pack_w(c, ws, fP(c, ix_down(l, r, 5)), 27, C, C);
    }
  }
  for (int l = 0; l < 6; ++l) {
    int C = PLANES[l];
    c.unin[l]     = pack_w(c, ws, fP(c, ix_up(l, 0, 4)), 1, 2 * C, C);
    c.uw[l][0][0] = pack_w(c, ws, fP(c, ix_up(l, 0, 5)), 27, 2 * C, C);
    c.uw[l][0][1] = pack_w(c, ws, fP(c, ix_up(l, 0, 6)), 27, C, C);
    c.uw[l][1][0] = pack_w(c, ws, fP(c, ix_up(l, 1, 4)), 27, C, C);
    c.uw[l][1][1] = pack_w(c, ws, fP(c, ix_up(l, 1, 5)), 27, C, C);
  }
  c.lw = pack_w(c, ws, fP(c, ix_lw()), 1, PLANES[0], 20);

  // ---- conv0: feats (N0 x 3, padded to 32) -> X[0] (N0 x 32) ----
  {
    long long total = (long long)N0 * 32;
    k_cvt16<<<(unsigned)((total + 255) / 256), 256, 0, stream>>>(fP(c, 0), N0, 3, 32, c.T16b);
    smconv(c, c.T16b, 32, c.nbr[0], 27, c.conv0, N0, PLANES[0],
           c.X[0], PLANES[0], 0, nullptr, 0, 0, nullptr);
  }

  // ---- full U-Net ----
  unet(c, 0);

  // ---- final BN-ReLU + linear classifier (32 -> 20, +bias) ----
  bn(c, c.X[0], N0, PLANES[0], fP(c, ix_gf()), fP(c, ix_bf()), c.T16, 32, nullptr);
  smconv(c, c.T16, 32, nullptr, 1, c.lw, N0, 20,
         (float*)d_out, 20, 0, nullptr, 0, 0, fP(c, ix_lb()));
}
